// Glm4MoeMoE_25245817766049
// MI455X (gfx1250) — compile-verified, matrix operand order checked by reading, not run on hardware
//
#include <hip/hip_runtime.h>
#include <stdint.h>

#define HDIM 1024
#define IDIM 768
#define NEXP 16
#define NTOK 2048
#define TOPKX 4
#define NROWS (NTOK * TOPKX + NTOK)   // 8192 routed slot rows + 2048 shared rows
#define SHARED_OFF (NTOK * TOPKX)

typedef __attribute__((ext_vector_type(16))) __bf16 v16bf;
typedef __attribute__((ext_vector_type(8)))  float  v8f;

union Frag16 { uint4 q[2]; v16bf v; };

__device__ __forceinline__ unsigned short f2bf(float f) {
  unsigned int u = __float_as_uint(f);
  unsigned int r = u + 0x7FFFu + ((u >> 16) & 1u);   // round-to-nearest-even
  return (unsigned short)(r >> 16);
}
__device__ __forceinline__ unsigned int pack2bf(float a, float b) {
  return (unsigned int)f2bf(a) | ((unsigned int)f2bf(b) << 16);
}
__device__ __forceinline__ v8f zero8() {
  v8f z;
#pragma unroll
  for (int i = 0; i < 8; ++i) z[i] = 0.f;
  return z;
}
__device__ __forceinline__ float fast_sigmoid(float x) {
  return __builtin_amdgcn_rcpf(1.f + __expf(-x));
}
// async 16B global -> LDS copy (per lane), tracked by ASYNCcnt
__device__ __forceinline__ void async_copy16(unsigned int lds_off, const void* gptr) {
  asm volatile("global_load_async_to_lds_b128 %0, %1, off"
               :: "v"(lds_off), "v"(gptr) : "memory");
}
__device__ __forceinline__ void wait_async0() {
  asm volatile("s_wait_asynccnt 0x0" ::: "memory");
}

// ---------------- init: zero atomic counters ----------------
__global__ void init_kernel(int* counts, int* fill) {
  int t = threadIdx.x;
  if (t < NEXP) { counts[t] = 0; fill[t] = 0; }
}

// ---------------- x (fp32) -> xb (bf16 bits), 4 elems/thread ----------------
__global__ __launch_bounds__(256) void cvt_kernel(const float* __restrict__ x,
                                                  unsigned short* __restrict__ xb) {
  int i = (blockIdx.x * 256 + threadIdx.x) * 4;
  if (i < NTOK * HDIM) {
    float4 v = *(const float4*)(x + i);
    uint2 o;
    o.x = pack2bf(v.x, v.y);
    o.y = pack2bf(v.z, v.w);
    *(uint2*)(xb + i) = o;
  }
}

// ---------------- routing: one wave per token ----------------
__global__ __launch_bounds__(32) void route_kernel(
    const float* __restrict__ x, const float* __restrict__ gw,
    const float* __restrict__ gb, int* __restrict__ counts,
    int* __restrict__ texp, float* __restrict__ twgt) {
  const int n = blockIdx.x;
  const int lane = threadIdx.x;
  const float* xr = x + (size_t)n * HDIM;
  float xl[32];
#pragma unroll
  for (int i = 0; i < 32; ++i) xl[i] = xr[lane + 32 * i];
  float sc[NEXP];
#pragma unroll
  for (int e = 0; e < NEXP; ++e) {
    const float* g = gw + e * HDIM;
    float s = 0.f;
#pragma unroll
    for (int i = 0; i < 32; ++i) s += xl[i] * g[lane + 32 * i];
#pragma unroll
    for (int d = 16; d >= 1; d >>= 1) s += __shfl_xor(s, d, 32);
    sc[e] = fast_sigmoid(s);
  }
  if (lane == 0) {
    float sb[NEXP];
    for (int e = 0; e < NEXP; ++e) sb[e] = sc[e] + gb[e];
    float gs[4];
    for (int g = 0; g < 4; ++g) {       // sum of top-2 biased scores per group
      float m1 = -1e30f, m2 = -1e30f;
      for (int j = 0; j < 4; ++j) {
        float v = sb[g * 4 + j];
        if (v > m1) { m2 = m1; m1 = v; } else if (v > m2) { m2 = v; }
      }
      gs[g] = m1 + m2;
    }
    int g1 = 0;
    for (int g = 1; g < 4; ++g) if (gs[g] > gs[g1]) g1 = g;
    int g2 = -1;
    for (int g = 0; g < 4; ++g) { if (g == g1) continue; if (g2 < 0 || gs[g] > gs[g2]) g2 = g; }
    float masked[NEXP];
    for (int e = 0; e < NEXP; ++e) {
      int g = e >> 2;
      masked[e] = (g == g1 || g == g2) ? sb[e] : -1e30f;
    }
    int idx[TOPKX]; float wv[TOPKX]; float wsum = 0.f;
    for (int k = 0; k < TOPKX; ++k) {
      int best = 0; float bv = -1e31f;
      for (int e = 0; e < NEXP; ++e) {
        bool used = false;
        for (int j = 0; j < k; ++j) if (idx[j] == e) used = true;
        if (used) continue;
        if (masked[e] > bv) { bv = masked[e]; best = e; }
      }
      idx[k] = best; wv[k] = sc[best]; wsum += wv[k];
    }
    float inv = 2.5f * __builtin_amdgcn_rcpf(wsum + 1e-20f);
    for (int k = 0; k < TOPKX; ++k) {
      texp[n * TOPKX + k] = idx[k];
      twgt[n * TOPKX + k] = wv[k] * inv;
      atomicAdd(&counts[idx[k]], 1);
    }
  }
}

// ---------------- exclusive prefix over expert counts ----------------
__global__ void offs_kernel(const int* __restrict__ counts, int* __restrict__ offs) {
  if (threadIdx.x == 0) {
    int a = 0;
    for (int e = 0; e < NEXP; ++e) { offs[e] = a; a += counts[e]; }
    offs[NEXP] = a;
  }
}

// ---------------- scatter tokens into per-expert lists ----------------
__global__ __launch_bounds__(256) void fill_kernel(
    const int* __restrict__ texp, const float* __restrict__ twgt,
    const int* __restrict__ offs, int* __restrict__ fill,
    int* __restrict__ tok_list, float* __restrict__ tok_w, int* __restrict__ slot) {
  int n = blockIdx.x * 256 + threadIdx.x;
  if (n >= NTOK) return;
  for (int k = 0; k < TOPKX; ++k) {
    int e = texp[n * TOPKX + k];
    int pos = atomicAdd(&fill[e], 1);
    int row = offs[e] + pos;
    tok_list[row] = n;
    tok_w[row] = twgt[n * TOPKX + k];
    slot[n * TOPKX + k] = row;
  }
}

// ---------------- GEMM1: x @ {w1,w3}^T, fused SiLU(g)*u*wt -> hact(bf16) ----
// grid: (IDIM/128, 2048/64, 17)   block: 256 (8 waves), wave tile 32x32
__global__ __launch_bounds__(256) void gemm1_kernel(
    const unsigned short* __restrict__ xb,
    const float* __restrict__ w1, const float* __restrict__ w3,
    const float* __restrict__ ws1, const float* __restrict__ ws3,
    const int* __restrict__ counts, const int* __restrict__ offs,
    const int* __restrict__ tok_list, const float* __restrict__ tok_w,
    unsigned short* __restrict__ hact) {
  const int e = blockIdx.z, mtile = blockIdx.y, ntile = blockIdx.x;
  int off, cnt; const float *b1, *b3;
  if (e < NEXP) {
    off = offs[e]; cnt = counts[e];
    b1 = w1 + (size_t)e * IDIM * HDIM;
    b3 = w3 + (size_t)e * IDIM * HDIM;
  } else { off = SHARED_OFF; cnt = NTOK; b1 = ws1; b3 = ws3; }
  const int m0 = mtile * 64;
  if (m0 >= cnt) return;
  const int vcnt = cnt - m0;

  __shared__ unsigned short As[64 * 40];    // 80B pitch
  __shared__ unsigned short B1s[128 * 40];
  __shared__ unsigned short B2s[128 * 40];
  __shared__ float wts[64];
  __shared__ int rtok[64];

  const int tid = threadIdx.x, lane = tid & 31, wave = tid >> 5;
  const int wm = wave & 1, wn = wave >> 1;

  if (tid < 64) {
    int tk = -1; float w = 0.f;
    if (tid < vcnt) {
      if (e < NEXP) { tk = tok_list[off + m0 + tid]; w = tok_w[off + m0 + tid]; }
      else { tk = m0 + tid; w = 1.0f; }
    }
    rtok[tid] = tk; wts[tid] = w;
  }
  __syncthreads();

  // A-tile async-copy setup: thread -> (row, 16B chunk); invalid rows skipped
  const int ar = tid >> 2, ac = tid & 3;
  const unsigned int a_lds =
      (unsigned int)(size_t)(As + ar * 40 + ac * 8);   // LDS byte offset
  const int a_tk = rtok[ar];
  const unsigned short* a_src =
      (a_tk >= 0) ? (xb + (size_t)a_tk * HDIM + ac * 8) : (const unsigned short*)0;

  // B-tile setup: thread -> (row, 16-float half)
  const int nr = tid >> 1, hh = tid & 1;
  const int ncol = ntile * 128 + nr;
  const float* s1 = b1 + (size_t)ncol * HDIM + hh * 16;
  const float* s3 = b3 + (size_t)ncol * HDIM + hh * 16;
  uint4* d1 = (uint4*)(B1s + nr * 40 + hh * 16);
  uint4* d2 = (uint4*)(B2s + nr * 40 + hh * 16);

  v8f accg[2][2], accu[2][2];
#pragma unroll
  for (int i = 0; i < 2; ++i)
#pragma unroll
    for (int j = 0; j < 2; ++j) { accg[i][j] = zero8(); accu[i][j] = zero8(); }

  for (int k0 = 0; k0 < HDIM; k0 += 32) {
    __syncthreads();
    // A: async global->LDS (bypasses VGPRs, overlaps with B convert below)
    if (a_src) async_copy16(a_lds, a_src + k0);
    {   // B: 128 rows x 32 f32 -> bf16 (both w1 and w3)
      const float4* p1 = (const float4*)(s1 + k0);
      const float4* p3 = (const float4*)(s3 + k0);
      float4 f0 = p1[0], f1 = p1[1], f2 = p1[2], f3 = p1[3];
      float4 g0 = p3[0], g1 = p3[1], g2 = p3[2], g3 = p3[3];
      d1[0] = make_uint4(pack2bf(f0.x, f0.y), pack2bf(f0.z, f0.w),
                         pack2bf(f1.x, f1.y), pack2bf(f1.z, f1.w));
      d1[1] = make_uint4(pack2bf(f2.x, f2.y), pack2bf(f2.z, f2.w),
                         pack2bf(f3.x, f3.y), pack2bf(f3.z, f3.w));
      d2[0] = make_uint4(pack2bf(g0.x, g0.y), pack2bf(g0.z, g0.w),
                         pack2bf(g1.x, g1.y), pack2bf(g1.z, g1.w));
      d2[1] = make_uint4(pack2bf(g2.x, g2.y), pack2bf(g2.z, g2.w),
                         pack2bf(g3.x, g3.y), pack2bf(g3.z, g3.w));
    }
    wait_async0();
    __syncthreads();

    const int hs = lane >> 4, lr = lane & 15;
    Frag16 af[2], bf1[2], bf3[2];
#pragma unroll
    for (int mt = 0; mt < 2; ++mt) {
      const unsigned short* base = As + (wm * 32 + mt * 16 + lr) * 40;
      af[mt].q[0] = *(const uint4*)(base + hs * 8);
      af[mt].q[1] = *(const uint4*)(base + 16 + hs * 8);
    }
#pragma unroll
    for (int nt = 0; nt < 2; ++nt) {
      const unsigned short* base1 = B1s + (wn * 32 + nt * 16 + lr) * 40;
      const unsigned short* base3 = B2s + (wn * 32 + nt * 16 + lr) * 40;
      bf1[nt].q[0] = *(const uint4*)(base1 + hs * 8);
      bf1[nt].q[1] = *(const uint4*)(base1 + 16 + hs * 8);
      bf3[nt].q[0] = *(const uint4*)(base3 + hs * 8);
      bf3[nt].q[1] = *(const uint4*)(base3 + 16 + hs * 8);
    }
#pragma unroll
    for (int mt = 0; mt < 2; ++mt)
#pragma unroll
      for (int nt = 0; nt < 2; ++nt) {
        accg[mt][nt] = __builtin_amdgcn_wmma_f32_16x16x32_bf16(
            false, af[mt].v, false, bf1[nt].v, (short)0, accg[mt][nt], false, false);
        accu[mt][nt] = __builtin_amdgcn_wmma_f32_16x16x32_bf16(
            false, af[mt].v, false, bf3[nt].v, (short)0, accu[mt][nt], false, false);
      }
  }

  const int grow0 = off + m0;
  const int lr = lane & 15, hs = lane >> 4;
#pragma unroll
  for (int mt = 0; mt < 2; ++mt)
#pragma unroll
    for (int nt = 0; nt < 2; ++nt) {
      int Nc = ntile * 128 + wn * 32 + nt * 16 + lr;
#pragma unroll
      for (int v = 0; v < 8; ++v) {
        int Mr = wm * 32 + mt * 16 + v + 8 * hs;
        if (Mr < vcnt) {
          float g = accg[mt][nt][v];
          float u = accu[mt][nt][v];
          float hv = g * fast_sigmoid(g) * u * wts[Mr];   // SiLU via v_rcp
          hact[(size_t)(grow0 + Mr) * IDIM + Nc] = f2bf(hv);
        }
      }
    }
}

// ---------------- GEMM2: hact @ w2^T -> ypart (fp32) ----------------
// grid: (HDIM/128, 2048/64, 17)
__global__ __launch_bounds__(256) void gemm2_kernel(
    const unsigned short* __restrict__ hact,
    const float* __restrict__ w2, const float* __restrict__ ws2,
    const int* __restrict__ counts, const int* __restrict__ offs,
    float* __restrict__ ypart) {
  const int e = blockIdx.z, mtile = blockIdx.y, ntile = blockIdx.x;
  int off, cnt; const float* bw;
  if (e < NEXP) { off = offs[e]; cnt = counts[e]; bw = w2 + (size_t)e * HDIM * IDIM; }
  else { off = SHARED_OFF; cnt = NTOK; bw = ws2; }
  const int m0 = mtile * 64;
  if (m0 >= cnt) return;
  const int vcnt = cnt - m0;

  __shared__ unsigned short As[64 * 40];
  __shared__ unsigned short Bs[128 * 40];

  const int tid = threadIdx.x, lane = tid & 31, wave = tid >> 5;
  const int wm = wave & 1, wn = wave >> 1;

  const int ar = tid >> 2, ac = tid & 3;
  const unsigned int a_lds =
      (unsigned int)(size_t)(As + ar * 40 + ac * 8);
  const unsigned short* a_src =
      (ar < vcnt) ? (hact + (size_t)(off + m0 + ar) * IDIM + ac * 8)
                  : (const unsigned short*)0;

  const int nr = tid >> 1, hh = tid & 1;
  const int ncol = ntile * 128 + nr;
  const float* s = bw + (size_t)ncol * IDIM + hh * 16;
  uint4* d = (uint4*)(Bs + nr * 40 + hh * 16);

  v8f acc[2][2];
#pragma unroll
  for (int i = 0; i < 2; ++i)
#pragma unroll
    for (int j = 0; j < 2; ++j) acc[i][j] = zero8();

  for (int k0 = 0; k0 < IDIM; k0 += 32) {
    __syncthreads();
    if (a_src) async_copy16(a_lds, a_src + k0);
    {
      const float4* p = (const float4*)(s + k0);
      float4 f0 = p[0], f1 = p[1], f2 = p[2], f3 = p[3];
      d[0] = make_uint4(pack2bf(f0.x, f0.y), pack2bf(f0.z, f0.w),
                        pack2bf(f1.x, f1.y), pack2bf(f1.z, f1.w));
      d[1] = make_uint4(pack2bf(f2.x, f2.y), pack2bf(f2.z, f2.w),
                        pack2bf(f3.x, f3.y), pack2bf(f3.z, f3.w));
    }
    wait_async0();
    __syncthreads();

    const int hs = lane >> 4, lr = lane & 15;
    Frag16 af[2], bf[2];
#pragma unroll
    for (int mt = 0; mt < 2; ++mt) {
      const unsigned short* base = As + (wm * 32 + mt * 16 + lr) * 40;
      af[mt].q[0] = *(const uint4*)(base + hs * 8);
      af[mt].q[1] = *(const uint4*)(base + 16 + hs * 8);
    }
#pragma unroll
    for (int nt = 0; nt < 2; ++nt) {
      const unsigned short* base = Bs + (wn * 32 + nt * 16 + lr) * 40;
      bf[nt].q[0] = *(const uint4*)(base + hs * 8);
      bf[nt].q[1] = *(const uint4*)(base + 16 + hs * 8);
    }
#pragma unroll
    for (int mt = 0; mt < 2; ++mt)
#pragma unroll
      for (int nt = 0; nt < 2; ++nt)
        acc[mt][nt] = __builtin_amdgcn_wmma_f32_16x16x32_bf16(
            false, af[mt].v, false, bf[nt].v, (short)0, acc[mt][nt], false, false);
  }

  const int grow0 = off + m0;
  const int lr = lane & 15, hs = lane >> 4;
#pragma unroll
  for (int mt = 0; mt < 2; ++mt)
#pragma unroll
    for (int nt = 0; nt < 2; ++nt) {
      int Nc = ntile * 128 + wn * 32 + nt * 16 + lr;
#pragma unroll
      for (int v = 0; v < 8; ++v) {
        int Mr = wm * 32 + mt * 16 + v + 8 * hs;
        if (Mr < vcnt)
          ypart[(size_t)(grow0 + Mr) * HDIM + Nc] = acc[mt][nt][v];
      }
    }
}

// ---------------- combine: out = shared + sum of 4 routed slots ----------------
__global__ __launch_bounds__(256) void combine_kernel(
    const float* __restrict__ ypart, const int* __restrict__ slot,
    float* __restrict__ out) {
  int n = blockIdx.x;
  int h = threadIdx.x * 4;                 // 256 threads * float4 = 1024 cols
  int s0 = slot[n * 4 + 0], s1 = slot[n * 4 + 1];
  int s2 = slot[n * 4 + 2], s3 = slot[n * 4 + 3];
  float4 v = *(const float4*)(ypart + (size_t)(SHARED_OFF + n) * HDIM + h);
  float4 a = *(const float4*)(ypart + (size_t)s0 * HDIM + h);
  float4 b = *(const float4*)(ypart + (size_t)s1 * HDIM + h);
  float4 c = *(const float4*)(ypart + (size_t)s2 * HDIM + h);
  float4 d = *(const float4*)(ypart + (size_t)s3 * HDIM + h);
  v.x += a.x + b.x + c.x + d.x;
  v.y += a.y + b.y + c.y + d.y;
  v.z += a.z + b.z + c.z + d.z;
  v.w += a.w + b.w + c.w + d.w;
  *(float4*)(out + (size_t)n * HDIM + h) = v;
}

extern "C" void kernel_launch(void* const* d_in, const int* in_sizes, int n_in,
                              void* d_out, int out_size, void* d_ws, size_t ws_size,
                              hipStream_t stream) {
  const float* x   = (const float*)d_in[0];
  const float* gw  = (const float*)d_in[1];
  const float* gb  = (const float*)d_in[2];
  const float* w1  = (const float*)d_in[3];
  const float* w2  = (const float*)d_in[4];
  const float* w3  = (const float*)d_in[5];
  const float* ws1 = (const float*)d_in[6];
  const float* ws2 = (const float*)d_in[7];
  const float* ws3 = (const float*)d_in[8];
  float* out = (float*)d_out;
  (void)in_sizes; (void)n_in; (void)out_size; (void)ws_size;

  char* p = (char*)d_ws;
  auto alloc = [&](size_t bytes) -> char* {
    char* r = p;
    p += (bytes + 255) & ~(size_t)255;
    return r;
  };
  unsigned short* xb   = (unsigned short*)alloc((size_t)NTOK * HDIM * 2);
  int*   counts   = (int*)  alloc(64);
  int*   fill     = (int*)  alloc(64);
  int*   offs     = (int*)  alloc(128);
  int*   texp     = (int*)  alloc((size_t)NTOK * TOPKX * 4);
  float* twgt     = (float*)alloc((size_t)NTOK * TOPKX * 4);
  int*   tok_list = (int*)  alloc((size_t)NTOK * TOPKX * 4);
  float* tok_w    = (float*)alloc((size_t)NTOK * TOPKX * 4);
  int*   slot     = (int*)  alloc((size_t)NTOK * TOPKX * 4);
  unsigned short* hact = (unsigned short*)alloc((size_t)NROWS * IDIM * 2);
  float* ypart    = (float*)alloc((size_t)NROWS * HDIM * 4);

  init_kernel<<<1, 64, 0, stream>>>(counts, fill);
  cvt_kernel<<<(NTOK * HDIM) / 1024, 256, 0, stream>>>(x, xb);
  route_kernel<<<NTOK, 32, 0, stream>>>(x, gw, gb, counts, texp, twgt);
  offs_kernel<<<1, 32, 0, stream>>>(counts, offs);
  fill_kernel<<<NTOK / 256, 256, 0, stream>>>(texp, twgt, offs, fill,
                                              tok_list, tok_w, slot);
  gemm1_kernel<<<dim3(IDIM / 128, NTOK / 64, NEXP + 1), 256, 0, stream>>>(
      xb, w1, w3, ws1, ws3, counts, offs, tok_list, tok_w, hact);
  gemm2_kernel<<<dim3(HDIM / 128, NTOK / 64, NEXP + 1), 256, 0, stream>>>(
      hact, w2, ws2, counts, offs, ypart);
  combine_kernel<<<NTOK, 256, 0, stream>>>(ypart, slot, out);
}
